// GPRGNNNet_64441689309219
// MI455X (gfx1250) — matrix operand order repaired and checked by reading, not compiled, not run on hardware
//
#include <hip/hip_runtime.h>
#include <hip/hip_bf16.h>
#include <math.h>

#define NN     100000
#define NE     3200000
#define C_IN   512
#define C_HID  256
#define C_OUT  64
#define KHOPS  10

typedef __bf16 bf16;
typedef __attribute__((ext_vector_type(16))) __bf16 v16bf;
typedef __attribute__((ext_vector_type(8)))  __bf16 v8bf;
typedef __attribute__((ext_vector_type(8)))  float  v8f;

#define BM   32
#define LDX  (C_IN + 8)    // 520 bf16 row stride (16B-aligned, bank-conflict pad)
#define LDH  (C_HID + 8)   // 264 bf16 row stride

// ---------------------------------------------------------------------------
// Fused MLP: h = (relu(x @ w1^T + b1)) @ w2^T + b2   (bf16 WMMA, fp32 accum)
// One workgroup = 32 rows. 8 waves. Stage1: 2x16 tiles of 16x16. Stage2: 2x4.
// ---------------------------------------------------------------------------
__global__ __launch_bounds__(256) void mlp_kernel(
    const float* __restrict__ x, const bf16* __restrict__ w1b,
    const float* __restrict__ b1, const bf16* __restrict__ w2b,
    const float* __restrict__ b2, float* __restrict__ h_out)
{
    __shared__ __align__(16) bf16 xs[BM * LDX];   // 32x512 x-tile (bf16)
    __shared__ __align__(16) bf16 h1s[BM * LDH];  // 32x256 hidden tile (bf16)

    const int t    = threadIdx.x;
    const int lane = t & 31;
    const int wave = t >> 5;
    const int row0 = blockIdx.x * BM;
    const int ln   = lane & 15;   // N (or M for A) index inside tile
    const int hi   = lane >> 4;   // lane-half selector

    // ---- cooperative load of x tile, fp32 -> bf16 into LDS ----
    {
        const float4* xg = (const float4*)(x + (size_t)row0 * C_IN);
        #pragma unroll
        for (int i = 0; i < 16; ++i) {
            int idx = t + i * 256;          // 0..4095
            int r   = idx >> 7;             // /128 float4 per row
            int cv  = idx & 127;
            float4 v = xg[(size_t)r * 128 + cv];
            bf16* d = &xs[r * LDX + cv * 4];
            d[0] = (bf16)v.x; d[1] = (bf16)v.y; d[2] = (bf16)v.z; d[3] = (bf16)v.w;
        }
    }
    __syncthreads();

    // ---- stage 1: h1 = relu(x @ w1^T + b1), 32 tiles, 4 per wave ----
    #pragma unroll
    for (int j = 0; j < 4; ++j) {
        int tid = wave + 8 * j;
        int mt  = tid >> 4;        // 0..1
        int nt  = tid & 15;        // 0..15
        int arow = mt * 16 + ln;
        int bn   = nt * 16 + ln;
        float bias = b1[bn];
        v8f acc = {};
        #pragma unroll
        for (int ks = 0; ks < C_IN / 32; ++ks) {
            // A fragment from LDS: row=arow, K = ks*32 + {hi*16..hi*16+15}
            const bf16* ap = &xs[arow * LDX + ks * 32 + hi * 16];
            v8bf alo = *(const v8bf*)(ap);
            v8bf ahi = *(const v8bf*)(ap + 8);
            v16bf a;
            #pragma unroll
            for (int i = 0; i < 8; ++i) { a[i] = alo[i]; a[i + 8] = ahi[i]; }
            // B fragment: w1b[n][k], 16 contiguous bf16 per lane
            v16bf b = *(const v16bf*)(w1b + (size_t)bn * C_IN + ks * 32 + hi * 16);
            acc = __builtin_amdgcn_wmma_f32_16x16x32_bf16(
                      false, a, false, b, (short)0, acc, false, false);
        }
        #pragma unroll
        for (int r = 0; r < 8; ++r) {
            int row = mt * 16 + r + hi * 8;
            float v = acc[r] + bias;
            h1s[row * LDH + bn] = (bf16)fmaxf(v, 0.0f);
        }
    }
    __syncthreads();

    // ---- stage 2: h = h1 @ w2^T + b2, 8 tiles, 1 per wave ----
    {
        int mt  = wave >> 2;       // 0..1
        int nt2 = wave & 3;        // 0..3
        int arow = mt * 16 + ln;
        int bn   = nt2 * 16 + ln;
        float bias = b2[bn];
        v8f acc = {};
        #pragma unroll
        for (int ks = 0; ks < C_HID / 32; ++ks) {
            const bf16* ap = &h1s[arow * LDH + ks * 32 + hi * 16];
            v8bf alo = *(const v8bf*)(ap);
            v8bf ahi = *(const v8bf*)(ap + 8);
            v16bf a;
            #pragma unroll
            for (int i = 0; i < 8; ++i) { a[i] = alo[i]; a[i + 8] = ahi[i]; }
            v16bf b = *(const v16bf*)(w2b + (size_t)bn * C_HID + ks * 32 + hi * 16);
            acc = __builtin_amdgcn_wmma_f32_16x16x32_bf16(
                      false, a, false, b, (short)0, acc, false, false);
        }
        #pragma unroll
        for (int r = 0; r < 8; ++r) {
            int row = row0 + mt * 16 + r + hi * 8;
            h_out[(size_t)row * C_OUT + bn] = acc[r] + bias;
        }
    }
}

// ---------------------------------------------------------------------------
// small streaming kernels
// ---------------------------------------------------------------------------
__global__ void cvt_bf16_kernel(const float* __restrict__ s, bf16* __restrict__ d, int n) {
    int i = blockIdx.x * blockDim.x + threadIdx.x;
    if (i < n) d[i] = (bf16)s[i];
}

__global__ void deg_kernel(const int* __restrict__ dstI, float* __restrict__ deg) {
    int i = blockIdx.x * blockDim.x + threadIdx.x;
    if (i < NE) unsafeAtomicAdd(&deg[dstI[i]], 1.0f);
}

__global__ void dinv_kernel(const float* __restrict__ deg, float* __restrict__ dinv) {
    int i = blockIdx.x * blockDim.x + threadIdx.x;
    if (i < NN) dinv[i] = rsqrtf(deg[i] + 1.0f);   // +1 self-loop; always > 0
}

__global__ void norm_kernel(const int* __restrict__ srcI, const int* __restrict__ dstI,
                            const float* __restrict__ dinv, float* __restrict__ nrm) {
    int i = blockIdx.x * blockDim.x + threadIdx.x;
    if (i < NE) nrm[i] = dinv[srcI[i]] * dinv[dstI[i]];
}

// out = gamma[0]*h ; z = h
__global__ void init_kernel(const float* __restrict__ h, const float* __restrict__ gamma,
                            float* __restrict__ out, float* __restrict__ z) {
    int i = blockIdx.x * blockDim.x + threadIdx.x;   // float4 index
    if (i < NN * C_OUT / 4) {
        float g0 = gamma[0];
        float4 v = ((const float4*)h)[i];
        ((float4*)z)[i] = v;
        float4 o = { g0 * v.x, g0 * v.y, g0 * v.z, g0 * v.w };
        ((float4*)out)[i] = o;
    }
}

// z_next = dinv^2 * z   (self-loop term; also initializes z_next)
__global__ void selfinit_kernel(float* __restrict__ zn, const float* __restrict__ z,
                                const float* __restrict__ dinv) {
    int i = blockIdx.x * blockDim.x + threadIdx.x;   // float4 index
    if (i < NN * C_OUT / 4) {
        float d = dinv[i >> 4];                      // 16 float4 per row of 64
        float w = d * d;
        float4 v = ((const float4*)z)[i];
        float4 o = { w * v.x, w * v.y, w * v.z, w * v.w };
        ((float4*)zn)[i] = o;
    }
}

// z_next[dst] += norm_e * z[src]   (16 threads/edge, float4 gather, 4 atomics)
__global__ void scatter_kernel(const int* __restrict__ srcI, const int* __restrict__ dstI,
                               const float* __restrict__ nrm, const float* __restrict__ z,
                               float* __restrict__ zn) {
    int t = blockIdx.x * blockDim.x + threadIdx.x;
    int e = t >> 4;
    if (e < NE) {
        int q = t & 15;
        int s = srcI[e], d = dstI[e];
        float w = nrm[e];
        float4 v = ((const float4*)(z + (size_t)s * C_OUT))[q];
        float* p = zn + (size_t)d * C_OUT + q * 4;
        unsafeAtomicAdd(p + 0, w * v.x);
        unsafeAtomicAdd(p + 1, w * v.y);
        unsafeAtomicAdd(p + 2, w * v.z);
        unsafeAtomicAdd(p + 3, w * v.w);
    }
}

// out += gamma[k] * z_next
__global__ void axpy_kernel(float* __restrict__ out, const float* __restrict__ zn,
                            const float* __restrict__ gamma, int k) {
    int i = blockIdx.x * blockDim.x + threadIdx.x;
    if (i < NN * C_OUT / 4) {
        float g = gamma[k];
        float4 v = ((const float4*)zn)[i];
        float4 o = ((float4*)out)[i];
        o.x += g * v.x; o.y += g * v.y; o.z += g * v.z; o.w += g * v.w;
        ((float4*)out)[i] = o;
    }
}

// row-wise log_softmax over 64 channels: one wave32 per row, 2 ch/lane
__global__ __launch_bounds__(256) void lsm_kernel(float* __restrict__ out) {
    int lane = threadIdx.x & 31;
    int row  = blockIdx.x * 8 + (threadIdx.x >> 5);
    if (row >= NN) return;
    float* p = out + (size_t)row * C_OUT;
    float a = p[lane], b = p[lane + 32];
    float m = fmaxf(a, b);
    #pragma unroll
    for (int o = 16; o > 0; o >>= 1) m = fmaxf(m, __shfl_xor(m, o, 32));
    float s = __expf(a - m) + __expf(b - m);
    #pragma unroll
    for (int o = 16; o > 0; o >>= 1) s += __shfl_xor(s, o, 32);
    float lse = m + __logf(s);
    p[lane]      = a - lse;
    p[lane + 32] = b - lse;
}

// ---------------------------------------------------------------------------
extern "C" void kernel_launch(void* const* d_in, const int* in_sizes, int n_in,
                              void* d_out, int out_size, void* d_ws, size_t ws_size,
                              hipStream_t stream) {
    const float* x     = (const float*)d_in[0];
    const float* w1    = (const float*)d_in[1];
    const float* b1    = (const float*)d_in[2];
    const float* w2    = (const float*)d_in[3];
    const float* b2    = (const float*)d_in[4];
    const float* gamma = (const float*)d_in[5];
    const int*   ei    = (const int*)d_in[6];
    const int*   srcI  = ei;
    const int*   dstI  = ei + NE;
    float* out = (float*)d_out;

    char* ws = (char*)d_ws;
    size_t off = 0;
    auto alloc = [&](size_t bytes) -> void* {
        off = (off + 255) & ~(size_t)255;
        void* p = ws + off;
        off += bytes;
        return p;
    };
    float* h    = (float*)alloc((size_t)NN * C_OUT * 4);
    float* zA   = (float*)alloc((size_t)NN * C_OUT * 4);
    float* zB   = (float*)alloc((size_t)NN * C_OUT * 4);
    float* deg  = (float*)alloc((size_t)NN * 4);
    float* dinv = (float*)alloc((size_t)NN * 4);
    float* nrm  = (float*)alloc((size_t)NE * 4);
    bf16*  w1b  = (bf16*)alloc((size_t)C_HID * C_IN * 2);
    bf16*  w2b  = (bf16*)alloc((size_t)C_OUT * C_HID * 2);

    // --- normalization precompute ---
    hipMemsetAsync(deg, 0, (size_t)NN * 4, stream);
    cvt_bf16_kernel<<<(C_HID * C_IN + 255) / 256, 256, 0, stream>>>(w1, w1b, C_HID * C_IN);
    cvt_bf16_kernel<<<(C_OUT * C_HID + 255) / 256, 256, 0, stream>>>(w2, w2b, C_OUT * C_HID);
    deg_kernel<<<(NE + 255) / 256, 256, 0, stream>>>(dstI, deg);
    dinv_kernel<<<(NN + 255) / 256, 256, 0, stream>>>(deg, dinv);
    norm_kernel<<<(NE + 255) / 256, 256, 0, stream>>>(srcI, dstI, dinv, nrm);

    // --- fused WMMA MLP ---
    mlp_kernel<<<NN / BM, 256, 0, stream>>>(x, w1b, b1, w2b, b2, h);

    // --- GPR propagation ---
    const int vec_blocks = (NN * C_OUT / 4 + 255) / 256;   // 6250
    init_kernel<<<vec_blocks, 256, 0, stream>>>(h, gamma, out, zA);
    float* z  = zA;
    float* zn = zB;
    for (int k = 1; k <= KHOPS; ++k) {
        selfinit_kernel<<<vec_blocks, 256, 0, stream>>>(zn, z, dinv);
        scatter_kernel<<<(NE * 16 + 255) / 256, 256, 0, stream>>>(srcI, dstI, nrm, z, zn);
        axpy_kernel<<<vec_blocks, 256, 0, stream>>>(out, zn, gamma, k);
        float* tmp = z; z = zn; zn = tmp;
    }

    // --- log_softmax ---
    lsm_kernel<<<(NN + 7) / 8, 256, 0, stream>>>(out);
}
